// SelfAttention_41214506172922
// MI455X (gfx1250) — matrix operand order repaired and checked by reading, not compiled, not run on hardware
//
#include <hip/hip_runtime.h>
#include <stdint.h>

// ---------------------------------------------------------------------------
// Fused single-head self-attention for MI455X (gfx1250, wave32, WMMA).
//   B=4, S=4096, D_MODEL=512, D_OUT=64, fp32 in/out.
//   1) wprep : W^T -> bf16 workspace
//   2) proj  : Q/K/V projections via v_wmma_f32_16x16x32_bf16 (double-buffered)
//   3) attn  : flash attention, S^T = K*Q^T trick (score D-frag == PV A-frag),
//              online softmax in exp2 domain, double-buffered K/V/mask loads.
// Memory policy: mask + fp32 inputs are single-use streams -> non-temporal;
// bf16 Q/K/V^T (6MB) are re-read by every query tile -> default (L2-resident).
// bf16 packing via v_perm_b32 (1 VALU) after +0x8000 rounding bias, since
// __builtin_amdgcn_cvt_pk_bf16_f32 is unavailable in this toolchain.
// HBM floor ~360MB (mask-dominated) => ~15.5us @ 23.3TB/s.
// ---------------------------------------------------------------------------

typedef __attribute__((ext_vector_type(16))) __bf16 v16bf;
typedef __attribute__((ext_vector_type(8)))  float  v8f;
typedef __attribute__((ext_vector_type(4)))  unsigned u4v;
typedef __attribute__((ext_vector_type(4)))  int      i4v;
typedef __attribute__((ext_vector_type(4)))  float    f4v;

namespace {
constexpr int   kB  = 4;
constexpr int   kS  = 4096;
constexpr int   kDm = 512;
constexpr int   kDo = 64;
// 1/sqrt(D_OUT) * log2(e): softmax done in exp2 domain.
constexpr float kQScale = 0.125f * 1.44269504088896340736f;

union BF16Frag { v16bf v; u4v q[2]; unsigned w[8]; unsigned short h[16]; };
union F32Acc   { v8f   v; float f[8]; };
union BF16x8   { unsigned short h[8]; u4v q; };
} // namespace

// Cheap f32->bf16 (round-half-up; <=0.5ulp off RNE, fine at bf16 precision).
__device__ __forceinline__ unsigned short f2bf(float x) {
  return (unsigned short)((__builtin_bit_cast(unsigned, x) + 0x8000u) >> 16);
}

// Pack two f32 into one dword of bf16 (low = a, high = b) with a single
// v_perm_b32: selector 0x03020706 picks {b.hi16, a.hi16} from pool {S0=ua,S1=ub}.
__device__ __forceinline__ unsigned pk_bf16(float a, float b) {
  unsigned ua = __builtin_bit_cast(unsigned, a) + 0x8000u;
  unsigned ub = __builtin_bit_cast(unsigned, b) + 0x8000u;
  return __builtin_amdgcn_perm(ua, ub, 0x03020706u);
}

__device__ __forceinline__ v8f wmma_bf16(v16bf a, v16bf b, v8f c) {
  return __builtin_amdgcn_wmma_f32_16x16x32_bf16(false, a, false, b,
                                                 (short)0, c, false, false);
}

// --------------------------------------------------------------------------
// Kernel 1: transpose + convert the three 512x64 weight matrices to bf16
//           Wt[m][col][k]  (feature-major rows => contiguous B-fragments)
// --------------------------------------------------------------------------
__global__ void sa_wprep(const float* __restrict__ Wq,
                         const float* __restrict__ Wk,
                         const float* __restrict__ Wv,
                         unsigned short* __restrict__ Wt) {
  int idx = blockIdx.x * blockDim.x + threadIdx.x;
  if (idx >= 3 * kDo * kDm) return;
  int m   = idx / (kDo * kDm);
  int rem = idx % (kDo * kDm);
  int col = rem / kDm;
  int k   = rem % kDm;
  const float* W = (m == 0) ? Wq : ((m == 1) ? Wk : Wv);
  Wt[idx] = f2bf(W[k * kDo + col]);
}

// --------------------------------------------------------------------------
// Kernel 2: projections, double-buffered k-loop.
// --------------------------------------------------------------------------
struct ProjBlk {
  BF16Frag a;     // A fragment (input rows, bf16-converted)
  BF16Frag b[4];  // B fragments (W^T, 4 column tiles)
};

__device__ __forceinline__ void proj_load(ProjBlk& blk, int k0,
                                          const float* __restrict__ arow,
                                          const unsigned short* __restrict__ wbase,
                                          int c0, int koff, int lq) {
  // A: 16 fp32 (NT stream) -> bf16, halves {c0..c0+7, c0+16..c0+23}
  const f4v* p0 = (const f4v*)(arow + k0 + c0);
  const f4v* p1 = (const f4v*)(arow + k0 + c0 + 16);
  f4v x0 = __builtin_nontemporal_load(p0);
  f4v x1 = __builtin_nontemporal_load(p0 + 1);
  f4v y0 = __builtin_nontemporal_load(p1);
  f4v y1 = __builtin_nontemporal_load(p1 + 1);
  blk.a.w[0] = pk_bf16(x0[0], x0[1]);
  blk.a.w[1] = pk_bf16(x0[2], x0[3]);
  blk.a.w[2] = pk_bf16(x1[0], x1[1]);
  blk.a.w[3] = pk_bf16(x1[2], x1[3]);
  blk.a.w[4] = pk_bf16(y0[0], y0[1]);
  blk.a.w[5] = pk_bf16(y0[2], y0[3]);
  blk.a.w[6] = pk_bf16(y1[0], y1[1]);
  blk.a.w[7] = pk_bf16(y1[2], y1[3]);
#pragma unroll
  for (int n = 0; n < 4; ++n) {
    const u4v* wp = (const u4v*)(wbase + (size_t)(n * 16 + lq) * kDm + k0 + c0);
    blk.b[n].q[0] = wp[0];
    blk.b[n].q[1] = wp[2];
  }
}

__device__ __forceinline__ void proj_compute(const ProjBlk& blk, F32Acc acc[4]) {
#pragma unroll
  for (int n = 0; n < 4; ++n)
    acc[n].v = wmma_bf16(blk.a.v, blk.b[n].v, acc[n].v);
}

__global__ void sa_proj(const float* __restrict__ xq,
                        const float* __restrict__ xk,
                        const float* __restrict__ xv,
                        const float* __restrict__ bq,
                        const float* __restrict__ bk,
                        const float* __restrict__ bv,
                        const unsigned short* __restrict__ Wt,
                        unsigned short* __restrict__ qbuf,
                        unsigned short* __restrict__ kbuf,
                        unsigned short* __restrict__ vtbuf) {
  const int m    = blockIdx.y;
  const int wave = threadIdx.x >> 5;
  const int lane = threadIdx.x & 31;
  const int tile = blockIdx.x * 8 + wave;  // 0..1023 (16-row tiles)
  const int lq   = lane & 15;
  const int half = lane >> 4;
  const int c0   = half ? 8 : 0;
  const int koff = half ? 16 : 0;

  const float* src  = (m == 0) ? xq : ((m == 1) ? xk : xv);
  const float* bias = (m == 0) ? bq : ((m == 1) ? bk : bv);
  const float* arow = src + (size_t)(tile * 16 + lq) * kDm;
  const unsigned short* wbase = Wt + (size_t)m * kDo * kDm;

  F32Acc acc[4];
#pragma unroll
  for (int n = 0; n < 4; ++n) acc[n].v = (v8f){0,0,0,0,0,0,0,0};

  // Software pipeline: ping-pong two blocks so next loads overlap WMMAs.
  ProjBlk blkA, blkB;
  proj_load(blkA, 0, arow, wbase, c0, koff, lq);
  for (int k0 = 0; k0 < kDm; k0 += 64) {
    proj_load(blkB, k0 + 32, arow, wbase, c0, koff, lq);
    proj_compute(blkA, acc);
    proj_load(blkA, (k0 + 64) & (kDm - 1), arow, wbase, c0, koff, lq);
    proj_compute(blkB, acc);
  }

  // Epilogue: D fragment is lane=col, VGPR i = row (rbase+i).
  const int rbase = tile * 16 + (half ? 8 : 0);
#pragma unroll
  for (int n = 0; n < 4; ++n) {
    const int col = n * 16 + lq;
    const float bv_ = bias[col];
    if (m == 0) {            // Q: row-major bf16, pre-scaled for exp2 softmax
#pragma unroll
      for (int i = 0; i < 8; ++i)
        qbuf[(size_t)(rbase + i) * kDo + col] =
            f2bf((acc[n].f[i] + bv_) * kQScale);
    } else if (m == 1) {     // K: row-major bf16
#pragma unroll
      for (int i = 0; i < 8; ++i)
        kbuf[(size_t)(rbase + i) * kDo + col] = f2bf(acc[n].f[i] + bv_);
    } else {                 // V: transposed bf16 [b][col][s] -> one b128 store
      BF16x8 t;
#pragma unroll
      for (int i = 0; i < 4; ++i)
        ((unsigned*)t.h)[i] = pk_bf16(acc[n].f[2 * i] + bv_,
                                      acc[n].f[2 * i + 1] + bv_);
      const int batch = rbase >> 12;
      const int srow  = rbase & (kS - 1);
      *(u4v*)(vtbuf + ((size_t)(batch * kDo + col)) * kS + srow) = t.q;
    }
  }
}

// --------------------------------------------------------------------------
// Kernel 3: flash attention, double-buffered 32-key blocks.
// --------------------------------------------------------------------------
struct AttnBlk {
  BF16Frag ka[2][2];  // K A-fragments: [key-tile][feat-half]
  i4v      mlo[2], mhi[2];
  BF16Frag vf[4];     // V B-fragments: 4 feature tiles
};

__device__ __forceinline__ void attn_load(AttnBlk& blk, int kb0,
                                          const unsigned short* __restrict__ kbb,
                                          const unsigned short* __restrict__ vbb,
                                          const int* __restrict__ mrow,
                                          int c0, int kofh, int lq) {
#pragma unroll
  for (int t = 0; t < 2; ++t) {
    const unsigned short* krow = kbb + (size_t)(kb0 + 16 * t + lq) * kDo;
    blk.ka[t][0].q[0] = *(const u4v*)(krow + c0);
    blk.ka[t][0].q[1] = *(const u4v*)(krow + c0 + 16);
    blk.ka[t][1].q[0] = *(const u4v*)(krow + 32 + c0);
    blk.ka[t][1].q[1] = *(const u4v*)(krow + 32 + c0 + 16);
    const i4v* mp = (const i4v*)(mrow + kb0 + 16 * t);
    blk.mlo[t] = __builtin_nontemporal_load(mp);      // mask: single-use stream
    blk.mhi[t] = __builtin_nontemporal_load(mp + 1);
  }
#pragma unroll
  for (int n = 0; n < 4; ++n) {
    const u4v* vp = (const u4v*)(vbb + (size_t)(16 * n + lq) * kS + kb0 + c0);
    blk.vf[n].q[0] = vp[0];
    blk.vf[n].q[1] = vp[2];
  }
  // Keep the HBM-critical mask stream ahead of the pipeline.
  __builtin_prefetch((const void*)(mrow + kb0 + 1024), 0, 0);
}

__device__ __forceinline__ void attn_compute(const AttnBlk& blk,
                                             const BF16Frag qf[2], F32Acc o[4],
                                             float& m_r, float& l_r, int half) {
  // scores S^T for two 16-key tiles
  F32Acc s[2];
#pragma unroll
  for (int t = 0; t < 2; ++t) {
    v8f z = (v8f){0,0,0,0,0,0,0,0};
    z = wmma_bf16(blk.ka[t][0].v, qf[0].v, z);
    s[t].v = wmma_bf16(blk.ka[t][1].v, qf[1].v, z);
  }
  // mask
#pragma unroll
  for (int t = 0; t < 2; ++t) {
#pragma unroll
    for (int j = 0; j < 4; ++j) {
      if (blk.mlo[t][j] == 0) s[t].f[j]     = -1e30f;
      if (blk.mhi[t][j] == 0) s[t].f[4 + j] = -1e30f;
    }
  }
  // online softmax (exp2 domain)
  float bm = -1e30f;
#pragma unroll
  for (int t = 0; t < 2; ++t)
#pragma unroll
    for (int j = 0; j < 8; ++j) bm = fmaxf(bm, s[t].f[j]);
  bm = fmaxf(bm, __shfl_xor(bm, 16));  // pair the two half-rows
  const float mn    = fmaxf(m_r, bm);
  const float alpha = exp2f(m_r - mn);
  m_r = mn;

  BF16Frag pa;  // P in exactly the PV A-fragment layout
  float ps = 0.0f;
#pragma unroll
  for (int j = 0; j < 4; ++j) {
    float p0 = exp2f(s[0].f[2 * j] - mn);
    float p1 = exp2f(s[0].f[2 * j + 1] - mn);
    pa.w[j] = pk_bf16(p0, p1);
    ps += p0 + p1;
  }
#pragma unroll
  for (int j = 0; j < 4; ++j) {
    float p0 = exp2f(s[1].f[2 * j] - mn);
    float p1 = exp2f(s[1].f[2 * j + 1] - mn);
    pa.w[4 + j] = pk_bf16(p0, p1);
    ps += p0 + p1;
  }
  ps += __shfl_xor(ps, 16);
  l_r = l_r * alpha + ps;

  // broadcast alpha per query, rescale accumulators
  float av[8];
#pragma unroll
  for (int i = 0; i < 8; ++i) av[i] = __shfl(alpha, (half ? 8 : 0) + i);
#pragma unroll
  for (int n = 0; n < 4; ++n)
#pragma unroll
    for (int i = 0; i < 8; ++i) o[n].f[i] *= av[i];

  // P * V
#pragma unroll
  for (int n = 0; n < 4; ++n)
    o[n].v = wmma_bf16(pa.v, blk.vf[n].v, o[n].v);
}

__global__ void sa_attn(const int* __restrict__ mask,
                        const unsigned short* __restrict__ qbuf,
                        const unsigned short* __restrict__ kbuf,
                        const unsigned short* __restrict__ vtbuf,
                        float* __restrict__ out) {
  const int wave = threadIdx.x >> 5;
  const int lane = threadIdx.x & 31;
  const int qt   = blockIdx.x * 8 + wave;  // 0..1023 query tiles
  const int b    = qt >> 8;
  const int q0   = (qt & 255) << 4;
  const int lq   = lane & 15;
  const int half = lane >> 4;
  const int c0   = half ? 8 : 0;
  const int kofh = half ? 16 : 0;

  BF16Frag qf[2];
  {
    const unsigned short* qrow = qbuf + (size_t)(b * kS + q0 + lq) * kDo;
#pragma unroll
    for (int f = 0; f < 2; ++f) {
      const u4v* qp = (const u4v*)(qrow + 32 * f + c0);
      qf[f].q[0] = qp[0];
      qf[f].q[1] = qp[2];
    }
  }

  F32Acc o[4];
#pragma unroll
  for (int n = 0; n < 4; ++n) o[n].v = (v8f){0,0,0,0,0,0,0,0};
  float m_r = -INFINITY, l_r = 0.0f;

  const unsigned short* kbb  = kbuf  + (size_t)b * kS * kDo;
  const unsigned short* vbb  = vtbuf + (size_t)b * kDo * kS;
  const int*            mrow = mask + (size_t)(b * kS + q0 + lq) * kS + c0;

  // Software pipeline: ping-pong two 32-key blocks.
  AttnBlk blkA, blkB;
  attn_load(blkA, 0, kbb, vbb, mrow, c0, kofh, lq);
  for (int kb0 = 0; kb0 < kS; kb0 += 64) {
    attn_load(blkB, kb0 + 32, kbb, vbb, mrow, c0, kofh, lq);
    attn_compute(blkA, qf, o, m_r, l_r, half);
    attn_load(blkA, (kb0 + 64) & (kS - 1), kbb, vbb, mrow, c0, kofh, lq);
    attn_compute(blkB, qf, o, m_r, l_r, half);
  }

  // normalize + store fp32 output (NT: written once, never re-read)
  float inv[8];
#pragma unroll
  for (int i = 0; i < 8; ++i) {
    float lv = __shfl(l_r, (half ? 8 : 0) + i);
    inv[i] = 1.0f / lv;
  }
  const int rbase = q0 + (half ? 8 : 0);
#pragma unroll
  for (int n = 0; n < 4; ++n) {
    const int col = 16 * n + lq;
#pragma unroll
    for (int i = 0; i < 8; ++i)
      __builtin_nontemporal_store(
          o[n].f[i] * inv[i],
          out + (size_t)(b * kS + rbase + i) * kDo + col);
  }
}

// --------------------------------------------------------------------------
extern "C" void kernel_launch(void* const* d_in, const int* in_sizes, int n_in,
                              void* d_out, int out_size, void* d_ws,
                              size_t ws_size, hipStream_t stream) {
  const float* query = (const float*)d_in[0];
  const float* key   = (const float*)d_in[1];
  const float* value = (const float*)d_in[2];
  const int*   mask  = (const int*)d_in[3];
  const float* Wq    = (const float*)d_in[4];
  const float* bq    = (const float*)d_in[5];
  const float* Wk    = (const float*)d_in[6];
  const float* bk    = (const float*)d_in[7];
  const float* Wv    = (const float*)d_in[8];
  const float* bv    = (const float*)d_in[9];

  char* ws = (char*)d_ws;
  const size_t kWtBytes  = (size_t)3 * kDo * kDm * sizeof(unsigned short);
  const size_t kQkvBytes = (size_t)kB * kS * kDo * sizeof(unsigned short);
  unsigned short* Wt    = (unsigned short*)ws;
  unsigned short* qbuf  = (unsigned short*)(ws + kWtBytes);
  unsigned short* kbuf  = (unsigned short*)(ws + kWtBytes + kQkvBytes);
  unsigned short* vtbuf = (unsigned short*)(ws + kWtBytes + 2 * kQkvBytes);

  sa_wprep<<<dim3((3 * kDo * kDm + 255) / 256), dim3(256), 0, stream>>>(
      Wq, Wk, Wv, Wt);
  sa_proj<<<dim3(kB * kS / 16 / 8, 3), dim3(256), 0, stream>>>(
      query, key, value, bq, bk, bv, Wt, qbuf, kbuf, vtbuf);
  sa_attn<<<dim3(kB * kS / 16 / 8), dim3(256), 0, stream>>>(
      mask, qbuf, kbuf, vtbuf, (float*)d_out);
}
